// Mamba3Backbone_83537113907708
// MI455X (gfx1250) — matrix operand order repaired and checked by reading, not compiled, not run on hardware
//
#include <hip/hip_runtime.h>
#include <math.h>

// ---------------- problem constants ----------------
#define B_       2
#define T_       2048
#define NT_      512
#define DIM_     768
#define DEPTH_   8
#define MEL_     100
#define TDIM_    512
#define DINNER_  1536
#define NS_      128
#define PH_      64
#define QC_      64
#define NH_      24
#define CONVDIM_ 1792          // D_INNER + 2*NS
#define DINPROJ_ 3352          // 2*D_INNER + 2*NS + NH
#define FFI_     1536
#define ROWS_    (B_*T_)       // 4096
#define CN_      (T_/QC_)      // 32 chunks
#define KCIN_    712           // concat K before padding
#define KCPAD_   768           // zero-padded to multiple of 64

typedef __attribute__((ext_vector_type(16))) _Float16 v16h;
typedef __attribute__((ext_vector_type(8)))  float    v8f;
typedef __attribute__((ext_vector_type(4)))  _Float16 h4;
typedef __attribute__((ext_vector_type(4)))  float    f4;
typedef __attribute__((ext_vector_type(4)))  int      i4;

// ---------------- gfx1250 async global->LDS copy (ASYNCcnt path) -------
#if defined(__has_builtin)
#  if __has_builtin(__builtin_amdgcn_global_load_async_to_lds_b128) && \
      __has_builtin(__builtin_amdgcn_s_wait_asynccnt)
#    define USE_ASYNC_LDS 1
#  endif
#endif
#ifndef USE_ASYNC_LDS
#  define USE_ASYNC_LDS 0
#endif

// copy 16 bytes global -> LDS; async on gfx1250 toolchains that expose it
__device__ __forceinline__ void copy16_g2l(void* lds_dst, const void* gsrc)
{
#if USE_ASYNC_LDS
    __builtin_amdgcn_global_load_async_to_lds_b128(
        (__attribute__((address_space(1))) i4*)gsrc,
        (__attribute__((address_space(3))) i4*)lds_dst,
        0, 0);
#else
    *(f4*)lds_dst = *(const f4*)gsrc;
#endif
}
__device__ __forceinline__ void copy_g2l_join()
{
#if USE_ASYNC_LDS
    __builtin_amdgcn_s_wait_asynccnt(0);
#endif
}

// ---------------- device helpers ----------------
__device__ __forceinline__ float siluf(float x){ return x / (1.f + expf(-x)); }
__device__ __forceinline__ float softplusf(float x){ return (x > 20.f) ? x : log1pf(expf(x)); }
__device__ __forceinline__ float mishf(float x){ return x * tanhf(softplusf(x)); }
__device__ __forceinline__ float geluf(float x){
    return 0.5f * x * (1.f + tanhf(0.7978845608028654f * (x + 0.044715f * x * x * x)));
}

// =====================================================================
// WMMA GEMM: C[M,N] = epilogue(A[M,K] * Wt[K,N])
// Host guarantees: M % 128 == 0, K % 64 == 0, operands 16B aligned.
// Only N is guarded, via a block-uniform edge path.
// 128x128 macro tile, GK=64 per barrier (2 WMMA K-steps),
// 256 threads = 8 wave32 waves, wave -> 32x64 subtile = 2x4 fragments,
// 16 x v_wmma_f32_16x16x32_f16 per main-loop body.
// =====================================================================
#define GM 128
#define GN 128
#define GK 64

#define EPI_STORE 0   // C = acc (+bias)
#define EPI_GELU  1   // C = gelu(acc + bias)
#define EPI_RESID 2   // C = Cin + gate[b,n] * (acc (+bias))

__global__ __launch_bounds__(256)
void k_gemm(const float* __restrict__ A, int lda,
            const float* __restrict__ Wt, int ldw,
            float* __restrict__ C, int ldc,
            const float* __restrict__ bias,
            const float* __restrict__ gate, int gstride,
            const float* __restrict__ Cin,
            int M, int N, int K, int mode)
{
    __shared__ _Float16 As[GM][GK + 8];   // row-major A tile
    __shared__ _Float16 Bs[GN][GK + 8];   // transposed B tile: [n][k]

    const int m0 = blockIdx.y * GM;
    const int n0 = blockIdx.x * GN;
    const int tid   = threadIdx.x;
    const int lane  = tid & 31;
    const int wave  = tid >> 5;
    const int waveM = wave >> 1;      // 0..3 (32 rows each)
    const int waveN = wave & 1;       // 0..1 (64 cols each)
    const int lhalf = lane >> 4;      // 0/1
    const int lrow  = lane & 15;
    const bool nfull = (n0 + GN) <= N;   // block-uniform

    v8f acc[2][4];
    for (int i = 0; i < 2; i++)
        for (int j = 0; j < 4; j++)
            for (int e = 0; e < 8; e++) acc[i][j][e] = 0.f;

    const int ktiles = K / GK;
    for (int kt = 0; kt < ktiles; kt++) {
        const int k0 = kt * GK;

        // uniform-branch prefetch of next weight K-tile into L2
        if (nfull && (kt + 1 < ktiles)) {
            const float* p = &Wt[(size_t)(k0 + GK + (tid >> 3)) * ldw + n0 + (tid & 7) * 16];
            __builtin_prefetch(p, 0, 1);
        }

        // ---- stage A: 128x64 f32 -> f16, guard-free, float4 -> b64 stores
        for (int i = 0; i < 8; i++) {
            int idx = i * 256 + tid;          // (r, cg): 128 rows x 16 groups
            int r = idx >> 4, cg = idx & 15;
            f4 v = *(const f4*)(A + (size_t)(m0 + r) * lda + k0 + cg * 4);
            h4 hv; hv[0] = (_Float16)v[0]; hv[1] = (_Float16)v[1];
            hv[2] = (_Float16)v[2]; hv[3] = (_Float16)v[3];
            *(h4*)&As[r][cg * 4] = hv;
        }
        // ---- stage B transposed: Bs[n][k]; 4 strided loads -> one b64 store
        if (nfull) {
            for (int i = 0; i < 8; i++) {
                int idx = i * 256 + tid;      // (n, kg): 128 n x 16 groups
                int nc = idx >> 4, kg = idx & 15;
                const float* wp = Wt + (size_t)(k0 + kg * 4) * ldw + n0 + nc;
                h4 hv;
                hv[0] = (_Float16)wp[0];
                hv[1] = (_Float16)wp[(size_t)ldw];
                hv[2] = (_Float16)wp[(size_t)ldw * 2];
                hv[3] = (_Float16)wp[(size_t)ldw * 3];
                *(h4*)&Bs[nc][kg * 4] = hv;
            }
        } else {
            for (int i = 0; i < 8; i++) {
                int idx = i * 256 + tid;
                int nc = idx >> 4, kg = idx & 15;
                int gn = n0 + nc;
                h4 hv;
                if (gn < N) {
                    const float* wp = Wt + (size_t)(k0 + kg * 4) * ldw + gn;
                    hv[0] = (_Float16)wp[0];
                    hv[1] = (_Float16)wp[(size_t)ldw];
                    hv[2] = (_Float16)wp[(size_t)ldw * 2];
                    hv[3] = (_Float16)wp[(size_t)ldw * 3];
                } else {
                    hv[0] = hv[1] = hv[2] = hv[3] = (_Float16)0.f;
                }
                *(h4*)&Bs[nc][kg * 4] = hv;
            }
        }
        __syncthreads();

        // ---- two 32-deep WMMA steps per staged tile
        for (int ks = 0; ks < GK; ks += 32) {
            v16h af[2];
            for (int i = 0; i < 2; i++) {
                const _Float16* ap = &As[waveM * 32 + i * 16 + lrow][ks + lhalf * 8];
                v16h a;
                for (int e = 0; e < 8; e++) { a[e] = ap[e]; a[e + 8] = ap[e + 16]; }
                af[i] = a;
            }
            v16h bf[4];
            for (int j = 0; j < 4; j++) {
                const _Float16* bp = &Bs[waveN * 64 + j * 16 + lrow][ks + lhalf * 8];
                v16h b;
                for (int e = 0; e < 8; e++) { b[e] = bp[e]; b[e + 8] = bp[e + 16]; }
                bf[j] = b;
            }
            for (int i = 0; i < 2; i++)
                for (int j = 0; j < 4; j++)
                    acc[i][j] = __builtin_amdgcn_wmma_f32_16x16x32_f16(
                        false, af[i], false, bf[j], (short)0, acc[i][j], false, false);
        }
        __syncthreads();
    }

    // ---- epilogue; C/D layout: VGPR vi -> M = vi + 8*lhalf, N = lrow
    for (int i = 0; i < 2; i++) {
        for (int j = 0; j < 4; j++) {
            int nn = n0 + waveN * 64 + j * 16 + lrow;
            if (nn >= N) continue;
            for (int vi = 0; vi < 8; vi++) {
                int mm = m0 + waveM * 32 + i * 16 + lhalf * 8 + vi;
                float v = acc[i][j][vi];
                size_t ci = (size_t)mm * ldc + nn;
                if (mode == EPI_STORE) {
                    if (bias) v += bias[nn];
                    C[ci] = v;
                } else if (mode == EPI_GELU) {
                    C[ci] = geluf(v + bias[nn]);
                } else { // EPI_RESID
                    if (bias) v += bias[nn];
                    int bb = mm >> 11;                 // mm / T_
                    C[ci] = Cin[ci] + gate[(size_t)bb * gstride + nn] * v;
                }
            }
        }
    }
}

// zero-pad inp_w (712x768) into 768x768
__global__ __launch_bounds__(256)
void k_padw(const float* __restrict__ w, float* __restrict__ wp)
{
    int idx = blockIdx.x * 256 + threadIdx.x;
    if (idx >= KCPAD_ * DIM_) return;
    int r = idx / DIM_;
    wp[idx] = (r < KCIN_) ? w[idx] : 0.f;
}

// =====================================================================
// Time embedding: te(B,256) -> hidden = silu(te@t_w1+b1)
// =====================================================================
__global__ __launch_bounds__(256)
void k_time1(const float* __restrict__ timep, const float* __restrict__ w1,
             const float* __restrict__ b1, float* __restrict__ hidden)
{
    int b = blockIdx.x;
    __shared__ float te[256];
    {
        int k = threadIdx.x;
        int kk = k & 127;
        float fr = expf(-logf(10000.f) / 127.f * (float)kk);
        float ang = 1000.f * timep[b] * fr;
        te[k] = (k < 128) ? sinf(ang) : cosf(ang);
    }
    __syncthreads();
    for (int d = threadIdx.x; d < DIM_; d += 256) {
        float acc = b1[d];
        for (int k = 0; k < 256; k++) acc += te[k] * w1[k * DIM_ + d];
        hidden[b * DIM_ + d] = siluf(acc);
    }
}

// st = silu(hidden @ t_w2 + b2)
__global__ __launch_bounds__(256)
void k_time2(const float* __restrict__ hidden, const float* __restrict__ w2,
             const float* __restrict__ b2, float* __restrict__ st)
{
    int b = blockIdx.x;
    __shared__ float hb[DIM_];
    for (int i = threadIdx.x; i < DIM_; i += 256) hb[i] = hidden[b * DIM_ + i];
    __syncthreads();
    for (int d = threadIdx.x; d < DIM_; d += 256) {
        float acc = b2[d];
        for (int k = 0; k < DIM_; k++) acc += hb[k] * w2[k * DIM_ + d];
        st[b * DIM_ + d] = siluf(acc);
    }
}

// mod[l,b,j] = st[b] @ adaln_w[l] + adaln_b[l]
__global__ __launch_bounds__(256)
void k_mod(const float* __restrict__ st, const float* __restrict__ aw,
           const float* __restrict__ ab, float* __restrict__ mod)
{
    int idx = blockIdx.x * 256 + threadIdx.x;
    if (idx >= DEPTH_ * B_ * 6 * DIM_) return;
    int j = idx % (6 * DIM_);
    int b = (idx / (6 * DIM_)) % B_;
    int l = idx / (6 * DIM_ * B_);
    float acc = ab[l * 6 * DIM_ + j];
    const float* wl = aw + (size_t)l * DIM_ * 6 * DIM_;
    const float* sb = st + b * DIM_;
    for (int k = 0; k < DIM_; k++) acc += sb[k] * wl[(size_t)k * 6 * DIM_ + j];
    mod[((size_t)l * B_ + b) * 6 * DIM_ + j] = acc;
}

// fm[b,j] = st[b] @ fin_w + fin_b
__global__ __launch_bounds__(256)
void k_fm(const float* __restrict__ st, const float* __restrict__ fw,
          const float* __restrict__ fb, float* __restrict__ fm)
{
    int idx = blockIdx.x * 256 + threadIdx.x;
    if (idx >= B_ * 2 * DIM_) return;
    int j = idx % (2 * DIM_);
    int b = idx / (2 * DIM_);
    float acc = fb[j];
    const float* sb = st + b * DIM_;
    for (int k = 0; k < DIM_; k++) acc += sb[k] * fw[(size_t)k * 2 * DIM_ + j];
    fm[b * 2 * DIM_ + j] = acc;
}

// Acat[row, 0..767] = [x(100) | cond(100) | text_table[idx](512) | 0-pad(56)]
__global__ __launch_bounds__(256)
void k_concat(const float* __restrict__ x, const float* __restrict__ cond,
              const int* __restrict__ text, const float* __restrict__ tbl,
              float* __restrict__ Acat)
{
    int idx = blockIdx.x * 256 + threadIdx.x;
    if (idx >= ROWS_ * KCPAD_) return;
    int c = idx % KCPAD_;
    int row = idx / KCPAD_;
    int b = row / T_, t = row % T_;
    float v;
    if (c < MEL_)                    v = x[(size_t)row * MEL_ + c];
    else if (c < 2 * MEL_)           v = cond[(size_t)row * MEL_ + (c - MEL_)];
    else if (c < KCIN_) {
        int ti = (t < NT_) ? (text[b * NT_ + t] + 1) : 0;
        v = tbl[(size_t)ti * TDIM_ + (c - 2 * MEL_)];
    } else v = 0.f;                  // zero K-padding
    Acat[(size_t)row * KCPAD_ + c] = v;
}

// grouped positional conv (K=31, groups=16, 48 ch/group), fused Mish
__global__ __launch_bounds__(256)
void k_posconv(const float* __restrict__ in, const float* __restrict__ w,
               const float* __restrict__ bias, float* __restrict__ out, int addres)
{
    int tt = blockIdx.x, g = blockIdx.y, b = blockIdx.z;
    int t0 = tt * 64;
    __shared__ float s[94][48];  // rows t0-15 .. t0+78
    for (int i = threadIdx.x; i < 94 * 48; i += 256) {
        int r = i / 48, ci = i % 48;
        int t = t0 - 15 + r;
        s[r][ci] = (t >= 0 && t < T_) ? in[((size_t)(b * T_ + t)) * DIM_ + g * 48 + ci] : 0.f;
    }
    __syncthreads();
    for (int i = threadIdx.x; i < 64 * 48; i += 256) {
        int lt = i / 48, o = i % 48;
        int oc = g * 48 + o;
        float acc = bias[oc];
        for (int kw = 0; kw < 31; kw++)
            for (int ci = 0; ci < 48; ci++)
                acc += s[lt + kw][ci] * w[((size_t)kw * 48 + ci) * DIM_ + oc];
        float m = mishf(acc);
        size_t oi = ((size_t)(b * T_ + t0 + lt)) * DIM_ + oc;
        if (addres) out[oi] += m; else out[oi] = m;
    }
}

// per-row LayerNorm with adaLN modulation: out = ln(h)*(1+sc)+sh
__global__ __launch_bounds__(256)
void k_ln(const float* __restrict__ h, const float* __restrict__ modb, int mstride,
          int sc_off, int sh_off, float* __restrict__ out)
{
    int row = blockIdx.x;
    int b = row / T_;
    const float* hr = h + (size_t)row * DIM_;
    __shared__ float red[256];
    float s = 0.f;
    for (int i = threadIdx.x; i < DIM_; i += 256) s += hr[i];
    red[threadIdx.x] = s; __syncthreads();
    for (int o = 128; o > 0; o >>= 1) { if (threadIdx.x < o) red[threadIdx.x] += red[threadIdx.x + o]; __syncthreads(); }
    float mean = red[0] * (1.f / DIM_);
    __syncthreads();
    float v = 0.f;
    for (int i = threadIdx.x; i < DIM_; i += 256) { float d = hr[i] - mean; v += d * d; }
    red[threadIdx.x] = v; __syncthreads();
    for (int o = 128; o > 0; o >>= 1) { if (threadIdx.x < o) red[threadIdx.x] += red[threadIdx.x + o]; __syncthreads(); }
    float rs = rsqrtf(red[0] * (1.f / DIM_) + 1e-6f);
    const float* sc = modb + (size_t)b * mstride + sc_off;
    const float* sh = modb + (size_t)b * mstride + sh_off;
    for (int i = threadIdx.x; i < DIM_; i += 256)
        out[(size_t)row * DIM_ + i] = (hr[i] - mean) * rs * (1.f + sc[i]) + sh[i];
}

// dt[row,h] = softplus(zxb[row, DINNER+CONVDIM+h] + dt_bias[h])
__global__ __launch_bounds__(256)
void k_dt(const float* __restrict__ zxb, const float* __restrict__ dtb,
          float* __restrict__ dt)
{
    int idx = blockIdx.x * 256 + threadIdx.x;
    if (idx >= ROWS_ * NH_) return;
    int hh = idx % NH_, row = idx / NH_;
    float v = zxb[(size_t)row * DINPROJ_ + DINNER_ + CONVDIM_ + hh] + dtb[hh];
    dt[idx] = softplusf(v);
}

// causal depthwise conv K=4 + SiLU on xBC slice of zxbcdt
__global__ __launch_bounds__(256)
void k_conv4(const float* __restrict__ zxb, const float* __restrict__ cw,
             const float* __restrict__ cb, float* __restrict__ xbc)
{
    int idx = blockIdx.x * 256 + threadIdx.x;
    if (idx >= ROWS_ * CONVDIM_) return;
    int c = idx % CONVDIM_;
    int row = idx / CONVDIM_;
    int b = row / T_, t = row % T_;
    float acc = cb[c];
    for (int k = 0; k < 4; k++) {
        int tt = t - 3 + k;
        if (tt >= 0)
            acc += zxb[((size_t)(b * T_ + tt)) * DINPROJ_ + DINNER_ + c] * cw[c * 4 + k];
    }
    xbc[(size_t)row * CONVDIM_ + c] = siluf(acc);
}

// per-(b,c,h): acum[q] = cumsum_q(dt * A)
__global__ __launch_bounds__(256)
void k_acum(const float* __restrict__ dt, const float* __restrict__ A_log,
            float* __restrict__ acum)
{
    int idx = blockIdx.x * 256 + threadIdx.x;
    if (idx >= B_ * CN_ * NH_) return;
    int h = idx % NH_;
    int c = (idx / NH_) % CN_;
    int b = idx / (NH_ * CN_);
    float A = -expf(A_log[h]);
    float run = 0.f;
    float* ao = acum + (size_t)idx * QC_;
    for (int q = 0; q < QC_; q++) {
        run += dt[((size_t)(b * T_ + c * QC_ + q)) * NH_ + h] * A;
        ao[q] = run;
    }
}

// CB[q,s] = Cc[q,:] . Bc[s,:]   per (b,c) block  (async-staged Bc tile)
__global__ __launch_bounds__(256)
void k_cb(const float* __restrict__ xbc, float* __restrict__ CB)
{
    int bc = blockIdx.x;
    int b = bc / CN_, c = bc % CN_;
    __shared__ float Bcs[QC_][NS_ + 4];
    for (int i = threadIdx.x; i < QC_ * (NS_ / 4); i += 256) {
        int s = i / (NS_ / 4), ng = i % (NS_ / 4);
        copy16_g2l(&Bcs[s][ng * 4],
                   &xbc[((size_t)(b * T_ + c * QC_ + s)) * CONVDIM_ + DINNER_ + ng * 4]);
    }
    copy_g2l_join();
    __syncthreads();
    for (int i = threadIdx.x; i < QC_ * QC_; i += 256) {
        int q = i / QC_, s = i % QC_;
        const float* cp = &xbc[((size_t)(b * T_ + c * QC_ + q)) * CONVDIM_ + DINNER_ + NS_];
        float acc = 0.f;
        for (int n = 0; n < NS_; n++) acc += cp[n] * Bcs[s][n];
        CB[((size_t)bc * QC_ + q) * QC_ + s] = acc;
    }
}

// y_in[q,p] = sum_{s<=q} CB[q,s]*exp(ac[q]-ac[s])*xb[s,p]   per (b,c,h)
__global__ __launch_bounds__(256)
void k_ssd_yin(const float* __restrict__ xbc, const float* __restrict__ dt,
               const float* __restrict__ acum, const float* __restrict__ CB,
               float* __restrict__ y)
{
    int bc = blockIdx.x, h = blockIdx.y;
    int b = bc / CN_, c = bc % CN_;
    __shared__ float xb[QC_][PH_ + 1];
    __shared__ float CBs[QC_][QC_];
    __shared__ float ac[QC_];
    // CB tile is a pure contiguous copy -> async path
    for (int i = threadIdx.x; i < QC_ * (QC_ / 4); i += 256) {
        int q = i / (QC_ / 4), sg = i % (QC_ / 4);
        copy16_g2l(&CBs[q][sg * 4], &CB[((size_t)bc * QC_ + q) * QC_ + sg * 4]);
    }
    for (int i = threadIdx.x; i < QC_ * PH_; i += 256) {
        int s = i / PH_, p = i % PH_;
        size_t row = (size_t)(b * T_ + c * QC_ + s);
        xb[s][p] = dt[row * NH_ + h] * xbc[row * CONVDIM_ + h * PH_ + p];
    }
    if (threadIdx.x < QC_)
        ac[threadIdx.x] = acum[((size_t)(bc * NH_ + h)) * QC_ + threadIdx.x];
    copy_g2l_join();
    __syncthreads();
    for (int i = threadIdx.x; i < QC_ * PH_; i += 256) {
        int q = i / PH_, p = i % PH_;
        float aq = ac[q];
        float acc = 0.f;
        for (int s = 0; s <= q; s++)
            acc += CBs[q][s] * expf(aq - ac[s]) * xb[s][p];
        size_t row = (size_t)(b * T_ + c * QC_ + q);
        y[row * DINNER_ + h * PH_ + p] = acc;
    }
}

// S[p,n] = sum_q exp(ac[63]-ac[q]) * xb[q,p] * Bc[q,n]   per (b,c,h)
__global__ __launch_bounds__(256)
void k_ssd_S(const float* __restrict__ xbc, const float* __restrict__ dt,
             const float* __restrict__ acum, float* __restrict__ S)
{
    int bc = blockIdx.x, h = blockIdx.y;
    int b = bc / CN_, c = bc % CN_;
    __shared__ float xw[QC_][PH_ + 1];
    __shared__ float Bcs[QC_][NS_ + 4];
    __shared__ float ac[QC_];
    // Bc tile: pure copy -> async path
    for (int i = threadIdx.x; i < QC_ * (NS_ / 4); i += 256) {
        int q = i / (NS_ / 4), ng = i % (NS_ / 4);
        copy16_g2l(&Bcs[q][ng * 4],
                   &xbc[((size_t)(b * T_ + c * QC_ + q)) * CONVDIM_ + DINNER_ + ng * 4]);
    }
    if (threadIdx.x < QC_)
        ac[threadIdx.x] = acum[((size_t)(bc * NH_ + h)) * QC_ + threadIdx.x];
    __syncthreads();
    float alast = ac[QC_ - 1];
    for (int i = threadIdx.x; i < QC_ * PH_; i += 256) {
        int q = i / PH_, p = i % PH_;
        size_t row = (size_t)(b * T_ + c * QC_ + q);
        xw[q][p] = expf(alast - ac[q]) * dt[row * NH_ + h] * xbc[row * CONVDIM_ + h * PH_ + p];
    }
    copy_g2l_join();
    __syncthreads();
    size_t base = ((size_t)bc * NH_ + h) * PH_ * NS_;
    for (int i = threadIdx.x; i < PH_ * NS_; i += 256) {
        int p = i / NS_, n = i % NS_;
        float acc = 0.f;
        for (int q = 0; q < QC_; q++) acc += xw[q][p] * Bcs[q][n];
        S[base + i] = acc;
    }
}

// sequential scan over chunks; converts S buffer in-place to prev-state
__global__ __launch_bounds__(256)
void k_ssd_scan(const float* __restrict__ acum, float* __restrict__ S)
{
    int blk = blockIdx.x;
    int b = blk / NH_, h = blk % NH_;
    float reg[32];
    for (int j = 0; j < 32; j++) reg[j] = 0.f;
    for (int c = 0; c < CN_; c++) {
        int bc = b * CN_ + c;
        float at = expf(acum[((size_t)(bc * NH_ + h)) * QC_ + (QC_ - 1)]);
        size_t base = ((size_t)bc * NH_ + h) * PH_ * NS_;
        for (int j = 0; j < 32; j++) {
            size_t idx = base + threadIdx.x + j * 256;
            float s = S[idx];
            S[idx] = reg[j];                // prev (state before this chunk)
            reg[j] = at * reg[j] + s;
        }
    }
}

// y += exp(ac[q]) * (Cc[q,:] . prev[p,:]) + Dp[h]*xh[q,p]
__global__ __launch_bounds__(256)
void k_ssd_yout(const float* __restrict__ xbc, const float* __restrict__ acum,
                const float* __restrict__ S, const float* __restrict__ Dp,
                float* __restrict__ y)
{
    int bc = blockIdx.x, h = blockIdx.y;
    int b = bc / CN_, c = bc % CN_;
    __shared__ float prevs[PH_][NS_ + 4];
    __shared__ float ac[QC_];
    size_t base = ((size_t)bc * NH_ + h) * PH_ * NS_;
    // prev state: pure contiguous copy -> async path
    for (int i = threadIdx.x; i < PH_ * (NS_ / 4); i += 256) {
        int p = i / (NS_ / 4), ng = i % (NS_ / 4);
        copy16_g2l(&prevs[p][ng * 4], &S[base + (size_t)p * NS_ + ng * 4]);
    }
    if (threadIdx.x < QC_)
        ac[threadIdx.x] = acum[((size_t)(bc * NH_ + h)) * QC_ + threadIdx.x];
    copy_g2l_join();
    __syncthreads();
    float dp = Dp[h];
    for (int i = threadIdx.x; i < QC_ * PH_; i += 256) {
        int q = i / PH_, p = i % PH_;
        size_t row = (size_t)(b * T_ + c * QC_ + q);
        const float* cp = &xbc[row * CONVDIM_ + DINNER_ + NS_];
        float acc = 0.f;
        for (int n = 0; n < NS_; n++) acc += cp[n] * prevs[p][n];
        size_t yi = row * DINNER_ + h * PH_ + p;
        float xh = xbc[row * CONVDIM_ + h * PH_ + p];
        y[yi] = y[yi] + expf(ac[q]) * acc + dp * xh;
    }
}

// y = (y * silu(z)) * rsqrt(mean(sq)+1e-5) * rms_w    (per 1536-row)
__global__ __launch_bounds__(256)
void k_gaterms(float* __restrict__ y, const float* __restrict__ zxb,
               const float* __restrict__ rmsw)
{
    int row = blockIdx.x;
    float* yr = y + (size_t)row * DINNER_;
    const float* zr = zxb + (size_t)row * DINPROJ_;
    __shared__ float red[256];
    float loc[6];
    float s = 0.f;
    for (int j = 0; j < 6; j++) {
        int i = threadIdx.x + j * 256;
        float t = yr[i] * siluf(zr[i]);
        loc[j] = t;
        s += t * t;
    }
    red[threadIdx.x] = s; __syncthreads();
    for (int o = 128; o > 0; o >>= 1) { if (threadIdx.x < o) red[threadIdx.x] += red[threadIdx.x + o]; __syncthreads(); }
    float scale = rsqrtf(red[0] * (1.f / DINNER_) + 1e-5f);
    for (int j = 0; j < 6; j++) {
        int i = threadIdx.x + j * 256;
        yr[i] = loc[j] * scale * rmsw[i];
    }
}

// =====================================================================
// host orchestration
// =====================================================================
extern "C" void kernel_launch(void* const* d_in, const int* in_sizes, int n_in,
                              void* d_out, int out_size, void* d_ws, size_t ws_size,
                              hipStream_t stream)
{
    (void)in_sizes; (void)n_in; (void)out_size; (void)ws_size;

    const float* x        = (const float*)d_in[0];
    const float* cond     = (const float*)d_in[1];
    const float* timep    = (const float*)d_in[2];
    const float* tbl      = (const float*)d_in[3];
    const float* inp_w    = (const float*)d_in[4];
    const float* inp_b    = (const float*)d_in[5];
    const float* pos_w1   = (const float*)d_in[6];
    const float* pos_b1   = (const float*)d_in[7];
    const float* pos_w2   = (const float*)d_in[8];
    const float* pos_b2   = (const float*)d_in[9];
    const float* t_w1     = (const float*)d_in[10];
    const float* t_b1     = (const float*)d_in[11];
    const float* t_w2     = (const float*)d_in[12];
    const float* t_b2     = (const float*)d_in[13];
    const float* adaln_w  = (const float*)d_in[14];
    const float* adaln_b  = (const float*)d_in[15];
    const float* in_w     = (const float*)d_in[16];
    const float* conv_w   = (const float*)d_in[17];
    const float* conv_b   = (const float*)d_in[18];
    const float* dt_bias  = (const float*)d_in[19];
    const float* A_log    = (const float*)d_in[20];
    const float* Dp       = (const float*)d_in[21];
    const float* rms_w    = (const float*)d_in[22];
    const float* out_w    = (const float*)d_in[23];
    const float* ff_w1    = (const float*)d_in[24];
    const float* ff_b1    = (const float*)d_in[25];
    const float* ff_w2    = (const float*)d_in[26];
    const float* ff_b2    = (const float*)d_in[27];
    const float* fin_w    = (const float*)d_in[28];
    const float* fin_b    = (const float*)d_in[29];
    const float* proj_w   = (const float*)d_in[30];
    const float* proj_b   = (const float*)d_in[31];
    const int*   text     = (const int*)d_in[32];
    float* outp = (float*)d_out;

    // ---- workspace carve ----
    float* W = (float*)d_ws;
    size_t off = 0;
    auto alloc = [&](size_t n) { float* p = W + off; off += n; return p; };
    float* st     = alloc((size_t)B_ * DIM_);
    float* hidden = alloc((size_t)B_ * DIM_);
    float* mod    = alloc((size_t)DEPTH_ * B_ * 6 * DIM_);
    float* fm     = alloc((size_t)B_ * 2 * DIM_);
    float* inp_wp = alloc((size_t)KCPAD_ * DIM_);     // zero-padded inp_w
    float* h      = alloc((size_t)ROWS_ * DIM_);
    float* xn     = alloc((size_t)ROWS_ * DIM_);
    float* zxb    = alloc((size_t)ROWS_ * DINPROJ_);
    float* xbc    = alloc((size_t)ROWS_ * CONVDIM_);
    float* dtb    = alloc((size_t)ROWS_ * NH_);
    float* acum   = alloc((size_t)B_ * CN_ * NH_ * QC_);
    float* CB     = alloc((size_t)B_ * CN_ * QC_ * QC_);
    float* ybuf   = alloc((size_t)ROWS_ * DINNER_);
    float* Sbuf   = alloc((size_t)B_ * CN_ * NH_ * PH_ * NS_);
    float* t1     = alloc((size_t)ROWS_ * FFI_);
    float* Acat = zxb;   // alias: only used before first layer (4096*768 fits)
    float* p1   = ybuf;  // alias: pos-conv temp

    auto gemm = [&](const float* A, int lda, const float* Wt, int ldw,
                    float* C, int ldc, const float* bias,
                    const float* gate, int gstride, const float* Cin,
                    int M, int N, int K, int mode) {
        dim3 g((N + GN - 1) / GN, (M + GM - 1) / GM);
        k_gemm<<<g, 256, 0, stream>>>(A, lda, Wt, ldw, C, ldc, bias, gate, gstride, Cin, M, N, K, mode);
    };

    // ---- conditioning ----
    k_time1<<<B_, 256, 0, stream>>>(timep, t_w1, t_b1, hidden);
    k_time2<<<B_, 256, 0, stream>>>(hidden, t_w2, t_b2, st);
    k_mod<<<(DEPTH_ * B_ * 6 * DIM_ + 255) / 256, 256, 0, stream>>>(st, adaln_w, adaln_b, mod);
    k_fm<<<(B_ * 2 * DIM_ + 255) / 256, 256, 0, stream>>>(st, fin_w, fin_b, fm);

    // ---- input projection (K zero-padded 712 -> 768 on both operands) ----
    k_padw<<<(KCPAD_ * DIM_ + 255) / 256, 256, 0, stream>>>(inp_w, inp_wp);
    k_concat<<<((size_t)ROWS_ * KCPAD_ + 255) / 256, 256, 0, stream>>>(x, cond, text, tbl, Acat);
    gemm(Acat, KCPAD_, inp_wp, DIM_, h, DIM_, inp_b, nullptr, 0, nullptr,
         ROWS_, DIM_, KCPAD_, EPI_STORE);

    // ---- positional convs (fused Mish) ----
    k_posconv<<<dim3(T_ / 64, 16, B_), 256, 0, stream>>>(h, pos_w1, pos_b1, p1, 0);
    k_posconv<<<dim3(T_ / 64, 16, B_), 256, 0, stream>>>(p1, pos_w2, pos_b2, h, 1);

    // ---- layers ----
    for (int l = 0; l < DEPTH_; l++) {
        const float* modl = mod + (size_t)l * B_ * 6 * DIM_;
        // LN + adaLN (sh_msa=chunk0, sc_msa=chunk1)
        k_ln<<<ROWS_, 256, 0, stream>>>(h, modl, 6 * DIM_, 1 * DIM_, 0 * DIM_, xn);
        // in_proj
        gemm(xn, DIM_, in_w + (size_t)l * DIM_ * DINPROJ_, DINPROJ_, zxb, DINPROJ_,
             nullptr, nullptr, 0, nullptr, ROWS_, DINPROJ_, DIM_, EPI_STORE);
        // dt = softplus(... + dt_bias)
        k_dt<<<(ROWS_ * NH_ + 255) / 256, 256, 0, stream>>>(zxb, dt_bias + l * NH_, dtb);
        // causal dwconv K=4 + SiLU
        k_conv4<<<((size_t)ROWS_ * CONVDIM_ + 255) / 256, 256, 0, stream>>>(
            zxb, conv_w + (size_t)l * CONVDIM_ * 4, conv_b + (size_t)l * CONVDIM_, xbc);
        // SSD
        k_acum<<<(B_ * CN_ * NH_ + 255) / 256, 256, 0, stream>>>(dtb, A_log + l * NH_, acum);
        k_cb<<<B_ * CN_, 256, 0, stream>>>(xbc, CB);
        k_ssd_yin<<<dim3(B_ * CN_, NH_), 256, 0, stream>>>(xbc, dtb, acum, CB, ybuf);
        k_ssd_S<<<dim3(B_ * CN_, NH_), 256, 0, stream>>>(xbc, dtb, acum, Sbuf);
        k_ssd_scan<<<B_ * NH_, 256, 0, stream>>>(acum, Sbuf);
        k_ssd_yout<<<dim3(B_ * CN_, NH_), 256, 0, stream>>>(xbc, acum, Sbuf, Dp + l * NH_, ybuf);
        // gate with silu(z), RMS norm
        k_gaterms<<<ROWS_, 256, 0, stream>>>(ybuf, zxb, rms_w + (size_t)l * DINNER_);
        // h += g_msa * (y @ out_w)
        gemm(ybuf, DINNER_, out_w + (size_t)l * DINNER_ * DIM_, DIM_, h, DIM_,
             nullptr, modl + 2 * DIM_, 6 * DIM_, h, ROWS_, DIM_, DINNER_, EPI_RESID);
        // LN2 (sh_mlp=chunk3, sc_mlp=chunk4)
        k_ln<<<ROWS_, 256, 0, stream>>>(h, modl, 6 * DIM_, 4 * DIM_, 3 * DIM_, xn);
        // ff1 + gelu
        gemm(xn, DIM_, ff_w1 + (size_t)l * DIM_ * FFI_, FFI_, t1, FFI_,
             ff_b1 + (size_t)l * FFI_, nullptr, 0, nullptr, ROWS_, FFI_, DIM_, EPI_GELU);
        // h += g_mlp * (t1 @ ff_w2 + ff_b2)
        gemm(t1, FFI_, ff_w2 + (size_t)l * FFI_ * DIM_, DIM_, h, DIM_,
             ff_b2 + (size_t)l * DIM_, modl + 5 * DIM_, 6 * DIM_, h, ROWS_, DIM_, FFI_, EPI_RESID);
    }

    // ---- final modulated LN + projection ----
    k_ln<<<ROWS_, 256, 0, stream>>>(h, fm, 2 * DIM_, 0, DIM_, xn);
    gemm(xn, DIM_, proj_w, MEL_, outp, MEL_, proj_b, nullptr, 0, nullptr,
         ROWS_, MEL_, DIM_, EPI_STORE);
}
